// AttentionConv_71339406787062
// MI455X (gfx1250) — matrix operand order repaired and checked by reading, not compiled, not run on hardware
//
#include <hip/hip_runtime.h>

// ---- CDNA5 WMMA vector types ----
typedef __attribute__((ext_vector_type(16))) __bf16 v16bf;
typedef __attribute__((ext_vector_type(8)))  __bf16 v8bf;
typedef __attribute__((ext_vector_type(8)))  float  v8f;
typedef __attribute__((ext_vector_type(4)))  float  v4f;
typedef __attribute__((ext_vector_type(4)))  unsigned int u32x4;
typedef __attribute__((ext_vector_type(8)))  unsigned int u32x8;

// Shapes
#define B_   8
#define CIN  128
#define NPTS 1024
#define MNB  16
#define G_   8
#define L_   96
#define CG   12
#define MC   64
#define CM   8
#define NL_  32

// ---- LDS helpers: get a 32-bit LDS byte offset from a generic pointer ----
typedef __attribute__((address_space(3))) char lds_char;
static __device__ __forceinline__ unsigned lds_byte_off(const void* p) {
  return (unsigned)(unsigned long long)(lds_char*)p;
}

// ---- CDNA5 async copy: global -> LDS, 16B per lane, tracked by ASYNCcnt ----
static __device__ __forceinline__ void async_load_b128(unsigned lds_off,
                                                       const void* gptr) {
  asm volatile("global_load_async_to_lds_b128 %0, %1, off"
               :: "v"(lds_off), "v"(gptr) : "memory");
}
static __device__ __forceinline__ void wait_async0() {
  asm volatile("s_wait_asynccnt 0x0" ::: "memory");
}

// ---- CDNA5 Tensor Data Mover: 2D tile global -> LDS, tracked by TENSORcnt --
// D# per ISA ch.8: group0 = {count, lds_addr, global_addr, type=2},
// group1 = {data_size, tensor dims / tile dims / stride}. 2-group (2D) form.
static __device__ __forceinline__ void tdm_load_2d(unsigned lds_dst,
                                                   const void* gsrc,
                                                   unsigned tensor_d0,
                                                   unsigned tensor_d1,
                                                   unsigned tile_d0,
                                                   unsigned tile_d1,
                                                   unsigned stride_d0) {
  unsigned long long ga = (unsigned long long)gsrc;
  u32x4 g0;
  g0[0] = 1u;                                        // count=1 (user D#)
  g0[1] = lds_dst;                                   // lds_addr [63:32]
  g0[2] = (unsigned)(ga & 0xFFFFFFFFu);              // global_addr[31:0]
  g0[3] = (unsigned)((ga >> 32) & 0x01FFFFFFu)       // global_addr[56:32]
          | (2u << 30);                              // type = 2 ("image")
  u32x8 g1;
  g1[0] = (2u << 16);                                // data_size=2 -> 4 bytes
  g1[1] = (tensor_d0 & 0xFFFFu) << 16;               // tensor_dim0 [79:48]
  g1[2] = ((tensor_d0 >> 16) & 0xFFFFu)
          | ((tensor_d1 & 0xFFFFu) << 16);           // tensor_dim1 [111:80]
  g1[3] = ((tensor_d1 >> 16) & 0xFFFFu)
          | ((tile_d0 & 0xFFFFu) << 16);             // tile_dim0 [127:112]
  g1[4] = (tile_d1 & 0xFFFFu);                       // tile_dim1 [143:128]
  g1[5] = stride_d0;                                 // tensor_dim0_stride lo
  g1[6] = 0u;                                        // stride hi / dim1_stride
  g1[7] = 0u;
  asm volatile("tensor_load_to_lds %0, %1" :: "s"(g0), "s"(g1) : "memory");
}

// ---------------------------------------------------------------------------
// Kernel 0: prep — convert Wq/Wk/Wv to bf16 (layout [3][96][128]) and zero
// the score + BN-stat accumulators (must happen every launch).
// ---------------------------------------------------------------------------
__global__ __launch_bounds__(256) void k_prep(const float* __restrict__ Wq,
                                              const float* __restrict__ Wk,
                                              const float* __restrict__ Wv,
                                              __bf16* __restrict__ Wbf,
                                              float* __restrict__ score,
                                              float* __restrict__ stats) {
  int i = blockIdx.x * 256 + threadIdx.x;     // 65536 threads
  if (i < L_ * CIN) {
    Wbf[0 * 12288 + i] = (__bf16)Wq[i];
    Wbf[1 * 12288 + i] = (__bf16)Wk[i];
    Wbf[2 * 12288 + i] = (__bf16)Wv[i];
  }
  if (i < B_ * G_ * NPTS) score[i] = 0.0f;
  if (i < 64) stats[i] = 0.0f;
}

// ---------------------------------------------------------------------------
// Kernel 1: fused local attention.
// One workgroup = (batch b, 16 points). 8 waves, each wave does 12 of the
// 96 (o-tile, n) 16x16 output tiles for each of q/k/v via bf16 WMMA.
//   - X f32 tile (128 x 256, row stride 64KB) : TDM tensor_load_to_lds
//   - W bf16 (72KB)                            : global_load_async_to_lds_b128
//   - f32 -> bf16 transposed repack done LDS -> LDS once both DMAs have completed.
// ---------------------------------------------------------------------------
// LDS layout (dynamic, bytes):
//   [0,      73728)  Xs   bf16 [256][144]  (p-major, pad 16 halfs)
//   [73728, 152064)  Ws   bf16 [288][136]  (row pad 8 halfs -> 272B rows)
//   [152064,160256)  logits f32 [8][16][16]
//   [160256,168448)  att    f32 [8][16][16]
//   [168448,169472)  idxs   int [16][16]
//   [169472,300544)  Xf   f32  [128][256]  (TDM destination)
#define XS_OFF   0
#define WS_OFF   73728
#define LG_OFF   152064
#define AT_OFF   160256
#define IX_OFF   168448
#define XF_OFF   169472
#define SMEM1_SZ 300544

__global__ __launch_bounds__(256)
void k_local(const float* __restrict__ x, const int* __restrict__ idx,
             const __bf16* __restrict__ Wbf, float* __restrict__ score,
             float* __restrict__ out) {
  const int b    = blockIdx.y;
  const int n0   = blockIdx.x * 16;        // first point of this tile
  const int tid  = threadIdx.x;
  const int lane = tid & 31;
  const int wave = tid >> 5;
  const int hi   = lane >> 4;              // half-wave select
  const int l15  = lane & 15;

  extern __shared__ char smem[];
  __bf16* Xs     = (__bf16*)(smem + XS_OFF);
  __bf16* Ws     = (__bf16*)(smem + WS_OFF);
  float*  logits = (float*)(smem + LG_OFF);
  float*  att    = (float*)(smem + AT_OFF);
  int*    idxs   = (int*)(smem + IX_OFF);
  float*  Xf     = (float*)(smem + XF_OFF);

  const float* xb = x + (size_t)b * CIN * (NPTS * MNB) + (size_t)n0 * MNB;

  // ---- Wave 0: kick off TDM DMA of the f32 X tile into LDS (TENSORcnt) ----
  if (tid < 32) {
    tdm_load_2d(lds_byte_off(smem) + XF_OFF, xb,
                /*tensor_d0=*/NPTS * MNB, /*tensor_d1=*/CIN,
                /*tile_d0=*/256, /*tile_d1=*/CIN,
                /*stride_d0=*/NPTS * MNB);
  }

  // ---- All waves: async DMA of W into LDS (ASYNCcnt), 16B chunks ----
  {
    const unsigned ws_lds = lds_byte_off(smem) + WS_OFF;
    const char* wg = (const char*)Wbf;
    #pragma unroll
    for (int it = 0; it < 18; ++it) {            // 18*256 = 4608 chunks
      int chunk = it * 256 + tid;                // [3*96 rows][16 chunks]
      int r = chunk >> 4, k16 = chunk & 15;
      async_load_b128(ws_lds + r * 272 + k16 * 16, wg + chunk * 16);
    }
  }

  // ---- Overlap: zero logits, fetch neighbor ids ----
  for (int i = tid; i < 2048; i += 256) logits[i] = 0.0f;
  {
    int n = tid >> 4, m = tid & 15;                // 256 = 16x16 exactly
    idxs[tid] = idx[(size_t)b * (NPTS * MNB) + (size_t)(n0 + n) * MNB + m];
  }
  wait_async0();
  if (tid < 32) __builtin_amdgcn_s_wait_tensorcnt(0);
  __syncthreads();

  // ---- LDS->LDS repack: f32 [c][p] -> bf16 [p][c] (vectorized reads) ----
  #pragma unroll
  for (int it = 0; it < 32; ++it) {
    int lin = it * 1024 + tid * 4;                 // 32768 floats total
    int c = lin >> 8, p = lin & 255;
    v4f fl = *(const v4f*)&Xf[lin];
    #pragma unroll
    for (int r = 0; r < 4; ++r) Xs[(p + r) * 144 + c] = (__bf16)fl[r];
  }
  __syncthreads();

  // ---- GEMM phase: q,k,v tiles; keep v fragments for later ----
  v8f vkeep[12];
  #pragma unroll
  for (int j = 0; j < 12; ++j) {
    const int pair = wave + 8 * j;       // 0..95
    const int t  = pair >> 4;            // o-tile 0..5
    const int pt = pair & 15;            // local point (n) 0..15
    v8f aq = {}, ak = {}, av = {};
    #pragma unroll
    for (int ks = 0; ks < 4; ++ks) {     // K = 128 in steps of 32
      // B fragment: X tile, 32(c) x 16(p); lane = col, K = e + 16*hi.
      const __bf16* bp = &Xs[(16 * pt + l15) * 144 + 32 * ks + 16 * hi];
      v16bf bf = *(const v16bf*)bp;      // 32B aligned LDS vector load
      #pragma unroll
      for (int mat = 0; mat < 3; ++mat) {
        // A fragment: W tile (LDS), 16(o) x 32(c); lane = row;
        // K order: K = e + 16*hi (contiguous) => two 16B chunks.
        const __bf16* wr =
            Ws + (mat * 96 + 16 * t + l15) * 136 + 32 * ks + 16 * hi;
        v8bf lo = *(const v8bf*)wr;
        v8bf hh = *(const v8bf*)(wr + 8);
        v16bf af;
        #pragma unroll
        for (int e = 0; e < 8; ++e) { af[e] = lo[e]; af[e + 8] = hh[e]; }
        if (mat == 0)
          aq = __builtin_amdgcn_wmma_f32_16x16x32_bf16(false, af, false, bf,
                                                       (short)0, aq, false, false);
        else if (mat == 1)
          ak = __builtin_amdgcn_wmma_f32_16x16x32_bf16(false, af, false, bf,
                                                       (short)0, ak, false, false);
        else
          av = __builtin_amdgcn_wmma_f32_16x16x32_bf16(false, af, false, bf,
                                                       (short)0, av, false, false);
      }
    }
    vkeep[j] = av;
    // Grouped logits: att_logit[g][n][m] += q*k summed over the group's
    // 12 channels (groups straddle 16-row tiles -> LDS float atomics).
    #pragma unroll
    for (int i = 0; i < 8; ++i) {
      int o = 16 * t + i + 8 * hi;       // output channel 0..95
      int g = o / CG;
      atomicAdd(&logits[(g * 16 + pt) * 16 + l15], aq[i] * ak[i]);
    }
  }
  __syncthreads();

  // ---- Softmax over the 16 neighbors, one (g,n) row per thread ----
  if (tid < 128) {
    float* Lr = &logits[tid * 16];
    float mx = Lr[0];
    #pragma unroll
    for (int m = 1; m < 16; ++m) mx = fmaxf(mx, Lr[m]);
    float e[16], s = 0.0f;
    #pragma unroll
    for (int m = 0; m < 16; ++m) { e[m] = __expf(Lr[m] - mx); s += e[m]; }
    float inv = 1.0f / s;
    #pragma unroll
    for (int m = 0; m < 16; ++m) att[tid * 16 + m] = e[m] * inv;
  }
  __syncthreads();

  // ---- out_l = sum_m att * v (16-lane reduction inside half-waves) ----
  #pragma unroll
  for (int j = 0; j < 12; ++j) {
    const int pair = wave + 8 * j;
    const int t = pair >> 4, pt = pair & 15;
    #pragma unroll
    for (int i = 0; i < 8; ++i) {
      int o = 16 * t + i + 8 * hi;
      int g = o / CG;
      float v = vkeep[j][i] * att[(g * 16 + pt) * 16 + l15];
      v += __shfl_xor(v, 1, 16);
      v += __shfl_xor(v, 2, 16);
      v += __shfl_xor(v, 4, 16);
      v += __shfl_xor(v, 8, 16);
      if (l15 == 0)
        out[(size_t)b * (128 * NPTS) + (size_t)o * NPTS + n0 + pt] = v;
    }
  }

  // ---- scatter-sum of att into score, last-occurrence dedup (.set) ----
  for (int item = tid; item < 2048; item += 256) {
    int g = item >> 8, n = (item >> 4) & 15, m = item & 15;
    int jn = idxs[n * 16 + m];
    bool last = true;
    for (int m2 = m + 1; m2 < 16; ++m2)
      if (idxs[n * 16 + m2] == jn) last = false;
    if (last)
      atomicAdd(&score[((size_t)b * G_ + g) * NPTS + jn],
                att[(g * 16 + n) * 16 + m]);
  }
}

// ---------------------------------------------------------------------------
// Kernel 2: top-16 (sorted desc, ties -> lower index) per (b,g).
// ---------------------------------------------------------------------------
__global__ __launch_bounds__(256)
void k_topk(const float* __restrict__ score, int* __restrict__ topidx,
            float* __restrict__ topval) {
  __shared__ float sv[1024];
  __shared__ float rv[256];
  __shared__ int   ri[256];
  const int bg = blockIdx.x, tid = threadIdx.x;
  for (int i = tid; i < 1024; i += 256) sv[i] = score[(size_t)bg * 1024 + i];
  __syncthreads();
  for (int r = 0; r < 16; ++r) {
    float best = -3.4e38f; int bi = 1 << 30;
    for (int i = tid; i < 1024; i += 256) {
      float v = sv[i];
      if (v > best || (v == best && i < bi)) { best = v; bi = i; }
    }
    rv[tid] = best; ri[tid] = bi;
    __syncthreads();
    for (int s = 128; s > 0; s >>= 1) {
      if (tid < s) {
        float v2 = rv[tid + s]; int i2 = ri[tid + s];
        if (v2 > rv[tid] || (v2 == rv[tid] && i2 < ri[tid])) {
          rv[tid] = v2; ri[tid] = i2;
        }
      }
      __syncthreads();
    }
    if (tid == 0) {
      topidx[bg * 16 + r] = ri[0];
      topval[bg * 16 + r] = rv[0];
      sv[ri[0]] = -3.4e38f;
    }
    __syncthreads();
  }
}

// ---------------------------------------------------------------------------
// Kernel 3: non-local attention per (b,g). Only the 16 gathered columns of
// km/vm are ever needed. Writes km_sel/vm_sel (outputs 2,3) + out_nl.
// ---------------------------------------------------------------------------
__global__ __launch_bounds__(256)
void k_nonlocal(const float* __restrict__ absx, const float* __restrict__ Wmq,
                const float* __restrict__ Wmk, const float* __restrict__ Wmv,
                const int* __restrict__ topidx, const float* __restrict__ topval,
                float* __restrict__ outnl, float* __restrict__ out_kmsel,
                float* __restrict__ out_vmsel) {
  __shared__ float Wl[3 * 8 * 64];
  __shared__ float ksel[8 * 16];
  __shared__ float vsel[8 * 16];
  __shared__ int   tix[16];
  __shared__ float tvl[16];
  const int bg = blockIdx.x, b = bg >> 3, g = bg & 7, tid = threadIdx.x;

  for (int i = tid; i < 1536; i += 256) {
    int mat = i >> 9, c = (i >> 6) & 7, kq = i & 63;
    const float* W = (mat == 0) ? Wmq : (mat == 1) ? Wmk : Wmv;
    Wl[i] = W[(g * CM + c) * 64 + kq];
  }
  if (tid < 16) { tix[tid] = topidx[bg * 16 + tid]; tvl[tid] = topval[bg * 16 + tid]; }
  __syncthreads();

  if (tid < 128) {                       // (c,j) pairs: 8 x 16
    int c = tid >> 4, j = tid & 15, nj = tix[j];
    float sk = 0.0f, sm = 0.0f;
    for (int kq = 0; kq < 64; ++kq) {
      float a = absx[((size_t)b * 64 + kq) * NPTS + nj];
      sk += Wl[512 + c * 64 + kq] * a;   // km
      sm += Wl[1024 + c * 64 + kq] * a;  // vm
    }
    float sv2 = sm * tanhf(tvl[j]);
    ksel[c * 16 + j] = sk; vsel[c * 16 + j] = sv2;
    out_kmsel[(size_t)bg * 128 + c * 16 + j] = sk;
    out_vmsel[(size_t)bg * 128 + c * 16 + j] = sv2;
  }
  __syncthreads();

  for (int n = tid; n < NPTS; n += 256) {
    float qm[8] = {0, 0, 0, 0, 0, 0, 0, 0};
    for (int kq = 0; kq < 64; ++kq) {
      float a = absx[((size_t)b * 64 + kq) * NPTS + n];
      #pragma unroll
      for (int c = 0; c < 8; ++c) qm[c] += Wl[c * 64 + kq] * a;
    }
    float lg[16]; float mx = -3.4e38f;
    #pragma unroll
    for (int m = 0; m < 16; ++m) {
      float s = 0.0f;
      #pragma unroll
      for (int c = 0; c < 8; ++c) s += qm[c] * ksel[c * 16 + m];
      lg[m] = s; mx = fmaxf(mx, s);
    }
    float ssum = 0.0f;
    #pragma unroll
    for (int m = 0; m < 16; ++m) { lg[m] = __expf(lg[m] - mx); ssum += lg[m]; }
    float inv = 1.0f / ssum;
    float ov[8] = {0, 0, 0, 0, 0, 0, 0, 0};
    #pragma unroll
    for (int m = 0; m < 16; ++m) {
      float w = lg[m] * inv;
      #pragma unroll
      for (int c = 0; c < 8; ++c) ov[c] += w * vsel[c * 16 + m];
    }
    #pragma unroll
    for (int c = 0; c < 8; ++c)
      outnl[((size_t)b * MC + g * CM + c) * NPTS + n] = ov[c];
  }
}

// ---------------------------------------------------------------------------
// Kernel 4a: h = Wm2nl(32x64) x out_nl, accumulate BN batch stats.
// ---------------------------------------------------------------------------
__global__ __launch_bounds__(256)
void k_m2nl(const float* __restrict__ outnl, const float* __restrict__ Wm2nl,
            float* __restrict__ hbuf, float* __restrict__ stats) {
  __shared__ float W[NL_ * 64];
  __shared__ float s1[NL_], s2[NL_];
  const int b = blockIdx.y, tid = threadIdx.x;
  const int n = blockIdx.x * 256 + tid;
  for (int i = tid; i < NL_ * 64; i += 256) W[i] = Wm2nl[i];
  if (tid < NL_) { s1[tid] = 0.0f; s2[tid] = 0.0f; }
  __syncthreads();
  float h[NL_];
  #pragma unroll
  for (int o = 0; o < NL_; ++o) h[o] = 0.0f;
  for (int c = 0; c < 64; ++c) {
    float a = outnl[((size_t)b * MC + c) * NPTS + n];
    #pragma unroll
    for (int o = 0; o < NL_; ++o) h[o] += W[o * 64 + c] * a;
  }
  #pragma unroll
  for (int o = 0; o < NL_; ++o) {
    hbuf[((size_t)b * NL_ + o) * NPTS + n] = h[o];
    atomicAdd(&s1[o], h[o]);
    atomicAdd(&s2[o], h[o] * h[o]);
  }
  __syncthreads();
  if (tid < NL_) {
    atomicAdd(&stats[tid], s1[tid]);
    atomicAdd(&stats[NL_ + tid], s2[tid]);
  }
}

// ---------------------------------------------------------------------------
// Kernel 4b: BN normalize into out channels 96..127.
// ---------------------------------------------------------------------------
__global__ __launch_bounds__(256)
void k_bn(const float* __restrict__ hbuf, const float* __restrict__ stats,
          const float* __restrict__ gamma, const float* __restrict__ beta,
          float* __restrict__ out) {
  int e = blockIdx.x * 256 + threadIdx.x;           // 262144
  int b = e >> 15, o = (e >> 10) & 31, n = e & 1023;
  float mu  = stats[o] * (1.0f / 8192.0f);
  float var = stats[NL_ + o] * (1.0f / 8192.0f) - mu * mu;
  float y = gamma[o] * (hbuf[e] - mu) * rsqrtf(var + 1e-5f) + beta[o];
  out[(size_t)b * (128 * NPTS) + (size_t)(L_ + o) * NPTS + n] = y;
}

// ---------------------------------------------------------------------------
extern "C" void kernel_launch(void* const* d_in, const int* in_sizes, int n_in,
                              void* d_out, int out_size, void* d_ws, size_t ws_size,
                              hipStream_t stream) {
  const float* x     = (const float*)d_in[0];
  const float* absx  = (const float*)d_in[1];
  const int*   idx   = (const int*)d_in[2];
  const float* Wq    = (const float*)d_in[5];
  const float* Wk    = (const float*)d_in[6];
  const float* Wv    = (const float*)d_in[7];
  const float* Wmq   = (const float*)d_in[8];
  const float* Wmk   = (const float*)d_in[9];
  const float* Wmv   = (const float*)d_in[10];
  const float* Wm2nl = (const float*)d_in[11];
  const float* gamma = (const float*)d_in[12];
  const float* beta  = (const float*)d_in[13];
  float* out = (float*)d_out;

  char* ws = (char*)d_ws;
  __bf16* Wbf   = (__bf16*)(ws + 0);          //  73728 B
  float* score  = (float*)(ws + 73728);       // 262144 B
  int*   topidx = (int*)(ws + 335872);        //   8192 B
  float* topval = (float*)(ws + 344064);      //   8192 B
  float* outnl  = (float*)(ws + 352256);      // 2097152 B
  float* hbuf   = (float*)(ws + 2449408);     // 1048576 B
  float* stats  = (float*)(ws + 3497984);     //    256 B

  k_prep<<<256, 256, 0, stream>>>(Wq, Wk, Wv, Wbf, score, stats);

  k_local<<<dim3(64, 8), 256, SMEM1_SZ, stream>>>(x, idx, Wbf, score, out);

  k_topk<<<64, 256, 0, stream>>>(score, topidx, topval);

  k_nonlocal<<<64, 256, 0, stream>>>(absx, Wmq, Wmk, Wmv, topidx, topval,
                                     outnl, out + 1048576, out + 1056768);

  k_m2nl<<<dim3(4, 8), 256, 0, stream>>>(outnl, Wm2nl, hbuf, stats);

  k_bn<<<1024, 256, 0, stream>>>(hbuf, stats, gamma, beta, out);
}
